// TokenSetTransformer_59579786330692
// MI455X (gfx1250) — compile-verified, matrix operand order checked by reading, not run on hardware
//
#include <hip/hip_runtime.h>
#include <hip/hip_bf16.h>
#include <math.h>

typedef __attribute__((ext_vector_type(16))) _Float16 v16h;
typedef __attribute__((ext_vector_type(8)))  _Float16 v8h;
typedef __attribute__((ext_vector_type(8)))  float    v8f;

#define BB 16
#define TT 1024
#define CC 256
#define EE 512

// ---------------------------------------------------------------------------
// Weight convert+transpose: Wt[n*K + k] = (f16)W[k*N + n]
// Makes every WMMA B-operand two contiguous 16B loads per lane.
// ---------------------------------------------------------------------------
__global__ void wt_f16(const float* __restrict__ W, _Float16* __restrict__ Wt,
                       int K, int N)
{
  long e = (long)blockIdx.x * blockDim.x + threadIdx.x;
  if (e >= (long)K * N) return;
  int n = (int)(e % N);
  long k = e / N;
  Wt[(long)n * K + k] = (_Float16)W[e];
}

// Elementwise f32 -> f16 (row-major preserved)
__global__ void cvt_f16(const float* __restrict__ X, _Float16* __restrict__ Y, long n)
{
  long e = (long)blockIdx.x * blockDim.x + threadIdx.x;
  if (e < n) Y[e] = (_Float16)X[e];
}

// ---------------------------------------------------------------------------
// WMMA GEMM: Out[M,N] = A16[M,K] @ Wt16[N,K]^T (+bias) (+addsrc) (relu)
// One wave per 16x16 tile. Per k-step: 4 x b128 loads + 1 v_wmma.
// A-fragment lane layout (16x32 f16): halves {k0+half*8 ..+7} and
// {k0+16+half*8 ..+7} of row (row0 + lane&15)  -> two aligned 16B chunks.
// ---------------------------------------------------------------------------
__global__ void wmma_gemm16(const _Float16* __restrict__ A, const _Float16* __restrict__ Wt,
                            const float* __restrict__ bias, const float* __restrict__ addsrc,
                            float* __restrict__ Out, _Float16* __restrict__ Out16,
                            int M, int N, int K, int relu)
{
  int nt = blockIdx.x, mt = blockIdx.y;
  int lane = threadIdx.x & 31;
  int half = lane >> 4, l15 = lane & 15;
  int row0 = mt * 16, col0 = nt * 16;

  const _Float16* arow = A  + (long)(row0 + l15) * K + half * 8;
  const _Float16* brow = Wt + (long)(col0 + l15) * K + half * 8;

  v8f acc;
  float binit = bias ? bias[col0 + l15] : 0.0f;
#pragma unroll
  for (int r = 0; r < 8; ++r) acc[r] = binit;

  for (int k0 = 0; k0 < K; k0 += 32) {
    v8h a0 = *(const v8h*)(arow + k0);
    v8h a1 = *(const v8h*)(arow + k0 + 16);
    v8h b0 = *(const v8h*)(brow + k0);
    v8h b1 = *(const v8h*)(brow + k0 + 16);
    if (k0 + 32 < K) {                       // global_prefetch_b8 next K slice
      __builtin_prefetch(arow + k0 + 32, 0, 3);
      __builtin_prefetch(brow + k0 + 32, 0, 3);
    }
    v16h a = __builtin_shufflevector(a0, a1, 0,1,2,3,4,5,6,7,8,9,10,11,12,13,14,15);
    v16h b = __builtin_shufflevector(b0, b1, 0,1,2,3,4,5,6,7,8,9,10,11,12,13,14,15);
    acc = __builtin_amdgcn_wmma_f32_16x16x32_f16(false, a, false, b,
                                                 (short)0, acc, false, false);
  }

  int n = col0 + l15;
  if (Out16) {
#pragma unroll
    for (int r = 0; r < 8; ++r) {
      int m = row0 + r + 8 * half;
      float v = acc[r];
      if (relu) v = fmaxf(v, 0.0f);
      Out16[(long)m * N + n] = (_Float16)v;
    }
  } else if (addsrc) {
#pragma unroll
    for (int r = 0; r < 8; ++r) {
      int m = row0 + r + 8 * half;
      float v = acc[r];
      if (relu) v = fmaxf(v, 0.0f);
      Out[(long)m * N + n] = v + addsrc[(long)m * N + n];
    }
  } else {
#pragma unroll
    for (int r = 0; r < 8; ++r) {
      int m = row0 + r + 8 * half;
      float v = acc[r];
      if (relu) v = fmaxf(v, 0.0f);
      Out[(long)m * N + n] = v;
    }
  }
}

// ---------------------------------------------------------------------------
// Farthest point sampling: one block per batch, sequential K-step argmax scan.
// ---------------------------------------------------------------------------
__global__ void fps_kernel(const float* __restrict__ centers, int Np, int Kc,
                           int* __restrict__ out_idx)
{
  __shared__ float px[1024], py[1024], pz[1024], dd[1024];
  __shared__ float redv[256];
  __shared__ int   redi[256];
  __shared__ int   s_cur;
  int b = blockIdx.x, tid = threadIdx.x;

  for (int i = tid; i < Np; i += 256) {
    px[i] = centers[(long)(b * Np + i) * 3 + 0];
    py[i] = centers[(long)(b * Np + i) * 3 + 1];
    pz[i] = centers[(long)(b * Np + i) * 3 + 2];
    dd[i] = INFINITY;
  }
  if (tid == 0) s_cur = 0;
  __syncthreads();

  for (int it = 0; it < Kc; ++it) {
    int cur = s_cur;
    if (tid == 0) out_idx[b * Kc + it] = cur;
    float cx = px[cur], cy = py[cur], cz = pz[cur];
    float bv = -INFINITY; int bi = 0;
    for (int i = tid; i < Np; i += 256) {
      float dx = px[i] - cx, dy = py[i] - cy, dz = pz[i] - cz;
      float nd = fminf(dd[i], dx * dx + dy * dy + dz * dz);
      dd[i] = nd;
      if (nd > bv) { bv = nd; bi = i; }
    }
    redv[tid] = bv; redi[tid] = bi;
    __syncthreads();
    for (int s = 128; s > 0; s >>= 1) {
      if (tid < s) {
        if (redv[tid + s] > redv[tid] ||
            (redv[tid + s] == redv[tid] && redi[tid + s] < redi[tid])) {
          redv[tid] = redv[tid + s]; redi[tid] = redi[tid + s];
        }
      }
      __syncthreads();
    }
    if (tid == 0) s_cur = redi[0];
    __syncthreads();
  }
}

__global__ void gather_rows(const float* __restrict__ src, const int* __restrict__ idx,
                            float* __restrict__ dst, int Tq, int Tt, int Wd)
{
  long e = (long)blockIdx.x * blockDim.x + threadIdx.x;
  long total = (long)BB * Tq * Wd;
  if (e >= total) return;
  int c = (int)(e % Wd);
  long r = e / Wd;
  int q = (int)(r % Tq);
  int b = (int)(r / Tq);
  int s = idx[b * Tq + q];
  dst[e] = src[((long)b * Tt + s) * Wd + c];
}

__global__ void knn_kernel(const float* __restrict__ cq, const float* __restrict__ ct,
                           int Tq, int Tt, int k, int* __restrict__ nn)
{
  __shared__ float d2[1024];
  __shared__ float redv[256];
  __shared__ int   redi[256];
  int qi = blockIdx.x, b = qi / Tq, q = qi % Tq, tid = threadIdx.x;
  float qx = cq[(long)(b * Tq + q) * 3 + 0];
  float qy = cq[(long)(b * Tq + q) * 3 + 1];
  float qz = cq[(long)(b * Tq + q) * 3 + 2];

  for (int i = tid; i < Tt; i += 256) {
    float dx = ct[(long)(b * Tt + i) * 3 + 0] - qx;
    float dy = ct[(long)(b * Tt + i) * 3 + 1] - qy;
    float dz = ct[(long)(b * Tt + i) * 3 + 2] - qz;
    d2[i] = dx * dx + dy * dy + dz * dz;
  }
  __syncthreads();

  for (int j = 0; j < k; ++j) {
    float bv = INFINITY; int bi = 0;
    for (int i = tid; i < Tt; i += 256) {
      float v = d2[i];
      if (v < bv) { bv = v; bi = i; }
    }
    redv[tid] = bv; redi[tid] = bi;
    __syncthreads();
    for (int s = 128; s > 0; s >>= 1) {
      if (tid < s) {
        if (redv[tid + s] < redv[tid] ||
            (redv[tid + s] == redv[tid] && redi[tid + s] < redi[tid])) {
          redv[tid] = redv[tid + s]; redi[tid] = redi[tid + s];
        }
      }
      __syncthreads();
    }
    if (tid == 0) { nn[(long)(b * Tq + q) * k + j] = redi[0]; d2[redi[0]] = INFINITY; }
    __syncthreads();
  }
}

__global__ void bn_stats(const float* __restrict__ X1, int n1,
                         const float* __restrict__ X2, int n2, int Cw,
                         float* __restrict__ mean, float* __restrict__ var)
{
  __shared__ float s1[256], s2[256];
  int c = blockIdx.x, tid = threadIdx.x;
  float sum = 0.0f, sq = 0.0f;
  for (int r = tid; r < n1; r += 256) { float v = X1[(long)r * Cw + c]; sum += v; sq += v * v; }
  if (X2)
    for (int r = tid; r < n2; r += 256) { float v = X2[(long)r * Cw + c]; sum += v; sq += v * v; }
  s1[tid] = sum; s2[tid] = sq;
  __syncthreads();
  for (int s = 128; s > 0; s >>= 1) {
    if (tid < s) { s1[tid] += s1[tid + s]; s2[tid] += s2[tid + s]; }
    __syncthreads();
  }
  if (tid == 0) {
    float n = (float)(n1 + n2);
    float m = s1[0] / n;
    mean[c] = m;
    var[c] = s2[0] / n - m * m;
  }
}

__global__ void bn_apply(const float* __restrict__ X, float* __restrict__ Y,
                         long rows, int Cw, const float* __restrict__ mean,
                         const float* __restrict__ var, const float* __restrict__ g,
                         const float* __restrict__ bta, int relu)
{
  long e = (long)blockIdx.x * blockDim.x + threadIdx.x;
  if (e >= rows * Cw) return;
  int c = (int)(e % Cw);
  float v = (X[e] - mean[c]) * rsqrtf(var[c] + 1e-5f) * g[c] + bta[c];
  if (relu) v = fmaxf(v, 0.0f);
  Y[e] = v;
}

// ---------------------------------------------------------------------------
// Fused vector attention: one block (8 waves) per query.
// diff/h in LDS f16 (b128 ds loads for A-fragments), W1t/W2t f16 [N,K]
// (b128 global loads for B-fragments), sim in LDS f32, then per-channel
// softmax over the k neighbors and weighted V-gather.
// ---------------------------------------------------------------------------
__global__ void attention_kernel(const float* __restrict__ Q, const float* __restrict__ KV,
                                 const int* __restrict__ nn, const float* __restrict__ skip,
                                 const _Float16* __restrict__ W1t, const float* __restrict__ b1,
                                 const _Float16* __restrict__ W2t, const float* __restrict__ b2,
                                 float* __restrict__ Out, int Tq, int Tt, int k)
{
  extern __shared__ char smem[];
  _Float16* s_diff = (_Float16*)smem;                // k*CC f16
  _Float16* s_h    = s_diff + (long)k * CC;          // k*CC f16
  float*    s_sim  = (float*)(s_h + (long)k * CC);   // k*CC f32
  float*    s_q    = s_sim + (long)k * CC;           // CC
  int*      s_nn   = (int*)(s_q + CC);               // k

  int qi = blockIdx.x, b = qi / Tq;
  int tid = threadIdx.x;
  int wave = tid >> 5, lane = tid & 31;
  int half = lane >> 4, l15 = lane & 15;

  s_q[tid] = Q[(long)qi * CC + tid];
  if (tid < k) s_nn[tid] = nn[(long)qi * k + tid];
  __syncthreads();

  for (int e = tid; e < k * CC; e += 256) {
    int j = e / CC, c = e % CC;
    int t = s_nn[j];
    float kv = KV[((long)b * Tt + t) * (2 * CC) + c];
    s_diff[e] = (_Float16)(s_q[c] - kv);
  }
  __syncthreads();

  const int ntile = CC / 16;           // 16
  const int tot = (k >> 4) * ntile;    // 16/32/64 -> multiple of 8 waves

  // GEMM1: h = relu(diff @ W1 + b1)
  for (int tl = wave; tl < tot; tl += 8) {
    int mt = tl / ntile, nt2 = tl % ntile;
    const _Float16* arow = s_diff + (mt * 16 + l15) * CC + half * 8;
    const _Float16* brow = W1t + (long)(nt2 * 16 + l15) * CC + half * 8;
    v8f acc;
    float bini = b1[nt2 * 16 + l15];
#pragma unroll
    for (int r = 0; r < 8; ++r) acc[r] = bini;
    for (int k0 = 0; k0 < CC; k0 += 32) {
      v8h a0 = *(const v8h*)(arow + k0);
      v8h a1 = *(const v8h*)(arow + k0 + 16);
      v8h b0 = *(const v8h*)(brow + k0);
      v8h b1 = *(const v8h*)(brow + k0 + 16);
      v16h a = __builtin_shufflevector(a0, a1, 0,1,2,3,4,5,6,7,8,9,10,11,12,13,14,15);
      v16h bb = __builtin_shufflevector(b0, b1, 0,1,2,3,4,5,6,7,8,9,10,11,12,13,14,15);
      acc = __builtin_amdgcn_wmma_f32_16x16x32_f16(false, a, false, bb,
                                                   (short)0, acc, false, false);
    }
#pragma unroll
    for (int r = 0; r < 8; ++r)
      s_h[(mt * 16 + r + 8 * half) * CC + nt2 * 16 + l15] = (_Float16)fmaxf(acc[r], 0.0f);
  }
  __syncthreads();

  // GEMM2: sim = h @ W2 + b2
  for (int tl = wave; tl < tot; tl += 8) {
    int mt = tl / ntile, nt2 = tl % ntile;
    const _Float16* arow = s_h + (mt * 16 + l15) * CC + half * 8;
    const _Float16* brow = W2t + (long)(nt2 * 16 + l15) * CC + half * 8;
    v8f acc;
    float bini = b2[nt2 * 16 + l15];
#pragma unroll
    for (int r = 0; r < 8; ++r) acc[r] = bini;
    for (int k0 = 0; k0 < CC; k0 += 32) {
      v8h a0 = *(const v8h*)(arow + k0);
      v8h a1 = *(const v8h*)(arow + k0 + 16);
      v8h b0 = *(const v8h*)(brow + k0);
      v8h b1 = *(const v8h*)(brow + k0 + 16);
      v16h a = __builtin_shufflevector(a0, a1, 0,1,2,3,4,5,6,7,8,9,10,11,12,13,14,15);
      v16h bb = __builtin_shufflevector(b0, b1, 0,1,2,3,4,5,6,7,8,9,10,11,12,13,14,15);
      acc = __builtin_amdgcn_wmma_f32_16x16x32_f16(false, a, false, bb,
                                                   (short)0, acc, false, false);
    }
#pragma unroll
    for (int r = 0; r < 8; ++r)
      s_sim[(mt * 16 + r + 8 * half) * CC + nt2 * 16 + l15] = acc[r];
  }
  __syncthreads();

  // softmax over k per channel + weighted V sum + skip
  int c = tid;
  float mx = -INFINITY;
  for (int j = 0; j < k; ++j) mx = fmaxf(mx, s_sim[j * CC + c]);
  float sum = 0.0f;
  for (int j = 0; j < k; ++j) {
    float e2 = __expf(s_sim[j * CC + c] - mx);
    s_sim[j * CC + c] = e2;
    sum += e2;
  }
  float o = 0.0f;
  for (int j = 0; j < k; ++j) {
    int t = s_nn[j];
    float v = KV[((long)b * Tt + t) * (2 * CC) + CC + c];
    o += s_sim[j * CC + c] * v;
  }
  Out[(long)qi * CC + c] = o / sum + skip[(long)qi * CC + c];
}

__global__ void mean_pool(const float* __restrict__ X, float* __restrict__ g, int Tq)
{
  int e = blockIdx.x * blockDim.x + threadIdx.x;
  if (e >= BB * CC) return;
  int b = e / CC, c = e % CC;
  float s = 0.0f;
  for (int t = 0; t < Tq; ++t) s += X[((long)b * Tq + t) * CC + c];
  g[e] = s / (float)Tq;
}

__global__ void topk_norm(const float* __restrict__ G, float* __restrict__ Out)
{
  __shared__ float vals[EE];
  __shared__ float redv[256];
  __shared__ int   redi[256];
  __shared__ float s_thr, s_norm;
  int b = blockIdx.x, tid = threadIdx.x;

  for (int e = tid; e < EE; e += 256) vals[e] = G[b * EE + e];
  __syncthreads();

  for (int it = 0; it < 64; ++it) {
    float bv = -INFINITY; int bi = 0;
    for (int e = tid; e < EE; e += 256) {
      float v = vals[e];
      if (v > bv) { bv = v; bi = e; }
    }
    redv[tid] = bv; redi[tid] = bi;
    __syncthreads();
    for (int s = 128; s > 0; s >>= 1) {
      if (tid < s) {
        if (redv[tid + s] > redv[tid] ||
            (redv[tid + s] == redv[tid] && redi[tid + s] < redi[tid])) {
          redv[tid] = redv[tid + s]; redi[tid] = redi[tid + s];
        }
      }
      __syncthreads();
    }
    if (tid == 0) { s_thr = redv[0]; vals[redi[0]] = -INFINITY; }
    __syncthreads();
  }

  float thr = s_thr;
  float sq = 0.0f;
  for (int e = tid; e < EE; e += 256) {
    float v = G[b * EE + e];
    if (v >= thr) sq += v * v;
  }
  redv[tid] = sq;
  __syncthreads();
  for (int s = 128; s > 0; s >>= 1) {
    if (tid < s) redv[tid] += redv[tid + s];
    __syncthreads();
  }
  if (tid == 0) s_norm = fmaxf(sqrtf(redv[0]), 1e-12f);
  __syncthreads();

  for (int e = tid; e < EE; e += 256) {
    float v = G[b * EE + e];
    Out[b * EE + e] = (v >= thr) ? v / s_norm : 0.0f;
  }
}

// ---------------------------------------------------------------------------
extern "C" void kernel_launch(void* const* d_in, const int* in_sizes, int n_in,
                              void* d_out, int out_size, void* d_ws, size_t ws_size,
                              hipStream_t stream)
{
  (void)in_sizes; (void)n_in; (void)out_size; (void)ws_size;

  const float* tokens  = (const float*)d_in[0];
  const float* centers = (const float*)d_in[1];
  const float* lrfs    = (const float*)d_in[2];
  const float* wq      = (const float*)d_in[3];
  const float* wkv     = (const float*)d_in[4];
  const float* mlp_w1  = (const float*)d_in[5];
  const float* mlp_b1  = (const float*)d_in[6];
  const float* mlp_w2  = (const float*)d_in[7];
  const float* mlp_b2  = (const float*)d_in[8];
  const float* bn1_g   = (const float*)d_in[9];
  const float* bn1_b   = (const float*)d_in[10];
  const float* bn2_g   = (const float*)d_in[11];
  const float* bn2_b   = (const float*)d_in[12];
  const float* att_w1  = (const float*)d_in[13];
  const float* att_b1  = (const float*)d_in[14];
  const float* att_w2  = (const float*)d_in[15];
  const float* att_b2  = (const float*)d_in[16];
  const float* emb_w   = (const float*)d_in[17];
  const float* emb_b   = (const float*)d_in[18];
  const float* embn_g  = (const float*)d_in[19];
  const float* embn_b  = (const float*)d_in[20];

  char* ws = (char*)d_ws;
  size_t off = 0;
  auto alloc = [&](size_t bytes) -> char* {
    char* p = ws + off;
    off += (bytes + 255) & ~(size_t)255;
    return p;
  };

  float* cQ[2]  = { (float*)alloc((size_t)BB*512*3*4), (float*)alloc((size_t)BB*512*3*4) };
  float* lQ[2]  = { (float*)alloc((size_t)BB*512*9*4), (float*)alloc((size_t)BB*512*9*4) };
  int*   repIdx = (int*)alloc((size_t)BB*512*4);
  int*   nnIdx  = (int*)alloc((size_t)BB*512*64*4);
  float* tokQ   = (float*)alloc((size_t)BB*512*CC*4);
  float* skipB  = (float*)alloc((size_t)BB*512*CC*4);
  float* Qbuf   = (float*)alloc((size_t)BB*512*CC*4);
  float* KVbuf  = (float*)alloc((size_t)BB*TT*2*CC*4);
  float* tqA    = (float*)alloc((size_t)BB*512*CC*4);
  float* tqB    = (float*)alloc((size_t)BB*512*CC*4);
  float* tqF[2] = { (float*)alloc((size_t)BB*512*CC*4), (float*)alloc((size_t)BB*512*CC*4) };
  float* meanB  = (float*)alloc((size_t)EE*4);
  float* varB   = (float*)alloc((size_t)EE*4);
  float* gbuf   = (float*)alloc((size_t)BB*CC*4);
  float* gebuf  = (float*)alloc((size_t)BB*EE*4);
  float* gbnbuf = (float*)alloc((size_t)BB*EE*4);

  // f16 transposed weights [N,K] + f16 activation staging
  _Float16* wqT    = (_Float16*)alloc((size_t)3*CC*CC*2);
  _Float16* wkvT   = (_Float16*)alloc((size_t)3*2*CC*CC*2);
  _Float16* aw1T   = (_Float16*)alloc((size_t)3*CC*CC*2);
  _Float16* aw2T   = (_Float16*)alloc((size_t)3*CC*CC*2);
  _Float16* mw1T   = (_Float16*)alloc((size_t)3*2*CC*CC*2);
  _Float16* mw2T   = (_Float16*)alloc((size_t)3*CC*2*CC*2);
  _Float16* embT   = (_Float16*)alloc((size_t)EE*CC*2);
  _Float16* tokQ16 = (_Float16*)alloc((size_t)BB*512*CC*2);
  _Float16* tokT16 = (_Float16*)alloc((size_t)BB*TT*CC*2);
  _Float16* tqB16  = (_Float16*)alloc((size_t)BB*512*CC*2);
  _Float16* hid16  = (_Float16*)alloc((size_t)BB*512*2*CC*2);
  _Float16* g16    = (_Float16*)alloc((size_t)BB*CC*2);

  // --- one-time weight convert+transpose (few MB; stays L2-resident) ---
  for (int i = 0; i < 3; ++i) {
    long sCC = (long)CC * CC, s2C = (long)CC * 2 * CC;
    wt_f16<<<(int)((sCC + 255) / 256), 256, 0, stream>>>(wq + i * sCC, wqT + i * sCC, CC, CC);
    wt_f16<<<(int)((s2C + 255) / 256), 256, 0, stream>>>(wkv + i * s2C, wkvT + i * s2C, CC, 2 * CC);
    wt_f16<<<(int)((sCC + 255) / 256), 256, 0, stream>>>(att_w1 + i * sCC, aw1T + i * sCC, CC, CC);
    wt_f16<<<(int)((sCC + 255) / 256), 256, 0, stream>>>(att_w2 + i * sCC, aw2T + i * sCC, CC, CC);
    wt_f16<<<(int)((s2C + 255) / 256), 256, 0, stream>>>(mlp_w1 + i * s2C, mw1T + i * s2C, CC, 2 * CC);
    wt_f16<<<(int)((s2C + 255) / 256), 256, 0, stream>>>(mlp_w2 + i * s2C, mw2T + i * s2C, 2 * CC, CC);
  }
  {
    long sE = (long)CC * EE;
    wt_f16<<<(int)((sE + 255) / 256), 256, 0, stream>>>(emb_w, embT, CC, EE);
  }

  const int NUMC[3] = {512, 256, 128};
  const int NUMK[3] = {16, 32, 64};

  const float* cen_t = centers;
  const float* lrf_t = lrfs;
  const float* tok_t = tokens;
  int Tt = TT;

  for (int i = 0; i < 3; ++i) {
    int Tq = NUMC[i], k = NUMK[i], pp = i & 1;
    long nq = (long)BB * Tq;
    long nt = (long)BB * Tt;

    fps_kernel<<<BB, 256, 0, stream>>>(cen_t, Tt, Tq, repIdx);

    long gc = nq * 3, gl = nq * 9, gt = nq * CC;
    gather_rows<<<(int)((gc + 255) / 256), 256, 0, stream>>>(cen_t, repIdx, cQ[pp], Tq, Tt, 3);
    gather_rows<<<(int)((gl + 255) / 256), 256, 0, stream>>>(lrf_t, repIdx, lQ[pp], Tq, Tt, 9);
    gather_rows<<<(int)((gt + 255) / 256), 256, 0, stream>>>(tok_t, repIdx, tokQ, Tq, Tt, CC);

    knn_kernel<<<(int)nq, 256, 0, stream>>>(cQ[pp], cen_t, Tq, Tt, k, nnIdx);

    // skip = tokens_query BEFORE bn1
    hipMemcpyAsync(skipB, tokQ, (size_t)nq * CC * 4, hipMemcpyDeviceToDevice, stream);

    if (i > 0) {
      bn_stats<<<CC, 256, 0, stream>>>(tokQ, (int)nq, tok_t, (int)nt, CC, meanB, varB);
      bn_apply<<<(int)((nq * CC + 255) / 256), 256, 0, stream>>>(
          tokQ, tokQ, nq, CC, meanB, varB, bn1_g + i * CC, bn1_b + i * CC, 0);
      bn_apply<<<(int)((nt * CC + 255) / 256), 256, 0, stream>>>(
          tok_t, (float*)tok_t, nt, CC, meanB, varB, bn1_g + i * CC, bn1_b + i * CC, 0);
    }

    // stage activations to f16 for WMMA
    cvt_f16<<<(int)((nq * CC + 255) / 256), 256, 0, stream>>>(tokQ, tokQ16, nq * CC);
    cvt_f16<<<(int)((nt * CC + 255) / 256), 256, 0, stream>>>(tok_t, tokT16, nt * CC);

    // Q = tokens_query @ wq[i]
    wmma_gemm16<<<dim3(CC / 16, (int)(nq / 16)), 32, 0, stream>>>(
        tokQ16, wqT + (long)i * CC * CC, nullptr, nullptr, Qbuf, nullptr, (int)nq, CC, CC, 0);
    // KV = tokens_target @ wkv[i]
    wmma_gemm16<<<dim3(2 * CC / 16, (int)(nt / 16)), 32, 0, stream>>>(
        tokT16, wkvT + (long)i * 2 * CC * CC, nullptr, nullptr, KVbuf, nullptr, (int)nt, 2 * CC, CC, 0);

    size_t smemB = (size_t)k * CC * 2 * 2 + (size_t)k * CC * 4
                 + (size_t)CC * 4 + (size_t)k * 4;
    attention_kernel<<<(int)nq, 256, smemB, stream>>>(
        Qbuf, KVbuf, nnIdx, skipB,
        aw1T + (long)i * CC * CC, att_b1 + (long)i * CC,
        aw2T + (long)i * CC * CC, att_b2 + (long)i * CC,
        tqA, Tq, Tt, k);

    // bn2 then MLP with residual
    bn_stats<<<CC, 256, 0, stream>>>(tqA, (int)nq, nullptr, 0, CC, meanB, varB);
    bn_apply<<<(int)((nq * CC + 255) / 256), 256, 0, stream>>>(
        tqA, tqB, nq, CC, meanB, varB, bn2_g + i * CC, bn2_b + i * CC, 0);
    cvt_f16<<<(int)((nq * CC + 255) / 256), 256, 0, stream>>>(tqB, tqB16, nq * CC);

    // hidden written directly as f16 (no f32 round-trip)
    wmma_gemm16<<<dim3(2 * CC / 16, (int)(nq / 16)), 32, 0, stream>>>(
        tqB16, mw1T + (long)i * 2 * CC * CC, mlp_b1 + (long)i * 2 * CC, nullptr,
        nullptr, hid16, (int)nq, 2 * CC, CC, 1);
    wmma_gemm16<<<dim3(CC / 16, (int)(nq / 16)), 32, 0, stream>>>(
        hid16, mw2T + (long)i * CC * 2 * CC, mlp_b2 + (long)i * CC, tqA,
        tqF[pp], nullptr, (int)nq, CC, 2 * CC, 0);

    cen_t = cQ[pp];
    lrf_t = lQ[pp];
    tok_t = tqF[pp];
    Tt = Tq;
  }

  // global embedding head
  mean_pool<<<(BB * CC + 255) / 256, 256, 0, stream>>>(tok_t, gbuf, Tt);
  cvt_f16<<<(BB * CC + 255) / 256, 256, 0, stream>>>(gbuf, g16, (long)BB * CC);
  wmma_gemm16<<<dim3(EE / 16, BB / 16), 32, 0, stream>>>(
      g16, embT, emb_b, nullptr, gebuf, nullptr, BB, EE, CC, 0);
  bn_stats<<<EE, 256, 0, stream>>>(gebuf, BB, nullptr, 0, EE, meanB, varB);
  bn_apply<<<(BB * EE + 255) / 256, 256, 0, stream>>>(
      gebuf, gbnbuf, BB, EE, meanB, varB, embn_g, embn_b, 1);
  topk_norm<<<BB, 256, 0, stream>>>(gbnbuf, (float*)d_out);
}
